// MultiHeadAttentionPooling_51144470561338
// MI455X (gfx1250) — compile-verified, matrix operand order checked by reading, not run on hardware
//
#include <hip/hip_runtime.h>

#define B_  32
#define L_  2048
#define D_  1024
#define H_  16
#define DH_ 64

typedef __attribute__((ext_vector_type(16))) __bf16 v16bf;
typedef __attribute__((ext_vector_type(8)))  __bf16 v8bf;
typedef __attribute__((ext_vector_type(8)))  float  v8f;
typedef __attribute__((ext_vector_type(4)))  int    v4i;

#define GAS __attribute__((address_space(1)))
#define LAS __attribute__((address_space(3)))

#if __has_builtin(__builtin_amdgcn_global_load_async_to_lds_b128) && \
    __has_builtin(__builtin_amdgcn_s_wait_asynccnt)
#define ASYNC_LDS 1
#else
#define ASYNC_LDS 0
#endif

// workspace layout (float offsets)
#define WS_WSCORE   0                       // H*D = 16384
#define WS_Q0       16384                   // 1024
#define WS_Q        17408                   // 1024
#define WS_SCORES   18432                   // B*H*L = 1048576
#define WS_POOLEDX  (18432 + 1048576)       // B*H*D = 524288
#define WS_CTX      (WS_POOLEDX + 524288)   // B*D = 32768
#define WS_Y        (WS_CTX + 32768)        // B*D = 32768
// total = 1,656,832 floats ~= 6.6 MB

// ---------------- K1a: q0 = LayerNorm(symbol_queries[pos]) ----------------
__global__ void __launch_bounds__(256) k_q0(const float* __restrict__ sym,
                                            const float* __restrict__ qn_w,
                                            const float* __restrict__ qn_b,
                                            const int* __restrict__ pos_p,
                                            float* __restrict__ ws) {
    __shared__ float red[256];
    int t = threadIdx.x;
    int pos = *pos_p;
    const float* x = sym + (size_t)pos * D_;
    float xl[4], s0 = 0.f, s1 = 0.f;
    #pragma unroll
    for (int i = 0; i < 4; i++) { xl[i] = x[t*4 + i]; s0 += xl[i]; s1 += xl[i]*xl[i]; }
    red[t] = s0; __syncthreads();
    for (int s = 128; s > 0; s >>= 1) { if (t < s) red[t] += red[t+s]; __syncthreads(); }
    float mu = red[0] * (1.0f / D_);
    __syncthreads();
    red[t] = s1; __syncthreads();
    for (int s = 128; s > 0; s >>= 1) { if (t < s) red[t] += red[t+s]; __syncthreads(); }
    float var = red[0] * (1.0f / D_) - mu*mu;
    float rs = rsqrtf(var + 1e-5f);
    #pragma unroll
    for (int i = 0; i < 4; i++) {
        int e = t*4 + i;
        ws[WS_Q0 + e] = (xl[i] - mu) * rs * qn_w[e] + qn_b[e];
    }
}

// ---------------- K1b: q = q0 @ Wq^T + bq  (grid 16, 4 threads per row) ---
__global__ void __launch_bounds__(256) k_qproj(const float* __restrict__ in_w,
                                               const float* __restrict__ in_b,
                                               float* __restrict__ ws) {
    __shared__ float part[256];
    int t = threadIdx.x;
    int j = blockIdx.x * 64 + (t >> 2);
    int p = t & 3;
    const float* wrow = in_w + (size_t)j * D_ + p * 256;
    const float* q0   = ws + WS_Q0 + p * 256;
    float acc = 0.f;
    for (int e = 0; e < 256; e++) acc += wrow[e] * q0[e];
    part[t] = acc; __syncthreads();
    if (p == 0) ws[WS_Q + j] = part[t] + part[t+1] + part[t+2] + part[t+3] + in_b[j];
}

// ---------------- K1c: w_score[h,e] = sum_d q[h*64+d]*Wk[h*64+d, e] -------
__global__ void __launch_bounds__(256) k_wscore(const float* __restrict__ in_w,
                                                float* __restrict__ ws) {
    int idx = blockIdx.x * 256 + threadIdx.x;   // grid 64 -> 16384 outputs
    int h = idx >> 10, e = idx & 1023;
    const float* q  = ws + WS_Q + h * DH_;
    const float* wk = in_w + (size_t)(D_ + h*DH_) * D_ + e;
    float acc = 0.f;
    #pragma unroll 8
    for (int d = 0; d < DH_; d++) acc += q[d] * wk[(size_t)d * D_];
    ws[WS_WSCORE + idx] = acc;
}

// ---------------- K2: scores = x @ w_score^T * scale + q.bk (WMMA) --------
__global__ void __launch_bounds__(256) k_scores(const float* __restrict__ x,
                                                const float* __restrict__ in_b,
                                                float* __restrict__ ws) {
    __shared__ __align__(16) __bf16 lb[H_ * D_];   // 32 KB, [h][e] bf16
    __shared__ float sbias[H_];
    int t = threadIdx.x;
    int b  = blockIdx.x >> 4;     // grid = 32 * 16
    int lt = blockIdx.x & 15;     // 128-row tile

    for (int i = t; i < H_ * D_; i += 256) lb[i] = (__bf16)ws[WS_WSCORE + i];
    if (t < H_) {
        const float* q  = ws + WS_Q + t * DH_;
        const float* bk = in_b + D_ + t * DH_;
        float s = 0.f;
        #pragma unroll
        for (int d = 0; d < DH_; d++) s += q[d] * bk[d];
        sbias[t] = s;
    }
    __syncthreads();

    int w = t >> 5, lane = t & 31;
    int hi = lane >> 4, hd = lane & 15, row = lane & 15;
    const float*  arow = x + ((size_t)b * L_ + lt*128 + w*16 + row) * D_;
    const __bf16* brow = lb + hd * D_;

    v8f c = {};
    for (int kb = 0; kb < D_; kb += 32) {
        const float* a0 = arow + kb + hi*8;
        __builtin_prefetch(a0 + 256, 0, 1);       // global_prefetch_b8, ~1KB ahead
        float4 f0 = *(const float4*)(a0);
        float4 f1 = *(const float4*)(a0 + 4);
        float4 f2 = *(const float4*)(a0 + 16);
        float4 f3 = *(const float4*)(a0 + 20);
        v16bf av;
        av[0]=(__bf16)f0.x; av[1]=(__bf16)f0.y; av[2]=(__bf16)f0.z; av[3]=(__bf16)f0.w;
        av[4]=(__bf16)f1.x; av[5]=(__bf16)f1.y; av[6]=(__bf16)f1.z; av[7]=(__bf16)f1.w;
        av[8]=(__bf16)f2.x; av[9]=(__bf16)f2.y; av[10]=(__bf16)f2.z; av[11]=(__bf16)f2.w;
        av[12]=(__bf16)f3.x; av[13]=(__bf16)f3.y; av[14]=(__bf16)f3.z; av[15]=(__bf16)f3.w;
        v8bf blo = *(const v8bf*)(brow + kb + hi*8);
        v8bf bhi = *(const v8bf*)(brow + kb + 16 + hi*8);
        v16bf bv = __builtin_shufflevector(blo, bhi, 0,1,2,3,4,5,6,7,8,9,10,11,12,13,14,15);
        c = __builtin_amdgcn_wmma_f32_16x16x32_bf16(false, av, false, bv, (short)0, c, false, false);
    }
    const float scale = 0.125f;   // 1/sqrt(64)
    float sb = sbias[hd];
    size_t base = (size_t)(b * H_ + hd) * L_ + lt*128 + w*16 + hi*8;
    #pragma unroll
    for (int r = 0; r < 8; r++)
        ws[WS_SCORES + base + r] = c[r] * scale + sb;
}

// ---------------- K3: softmax over L per (b,h); write attn to d_out -------
__global__ void __launch_bounds__(256) k_softmax(const float* __restrict__ ws,
                                                 float* __restrict__ attn_out) {
    __shared__ float red[256];
    int t = threadIdx.x;
    size_t base = (size_t)blockIdx.x * L_;     // grid 512 = b*H+h
    const float* s = ws + WS_SCORES + base;
    float vals[8], m = -1e30f;
    #pragma unroll
    for (int i = 0; i < 8; i++) { vals[i] = s[t + 256*i]; m = fmaxf(m, vals[i]); }
    red[t] = m; __syncthreads();
    for (int st = 128; st > 0; st >>= 1) { if (t < st) red[t] = fmaxf(red[t], red[t+st]); __syncthreads(); }
    m = red[0]; __syncthreads();
    float sum = 0.f;
    #pragma unroll
    for (int i = 0; i < 8; i++) { vals[i] = __expf(vals[i] - m); sum += vals[i]; }
    red[t] = sum; __syncthreads();
    for (int st = 128; st > 0; st >>= 1) { if (t < st) red[t] += red[t+st]; __syncthreads(); }
    float inv = 1.0f / red[0];
    #pragma unroll
    for (int i = 0; i < 8; i++) attn_out[base + t + 256*i] = vals[i] * inv;
}

// -- K5: pooled_x[b,h,:] = attn[b,h,:] @ x[b]  (WMMA, async double-buffer) --
__global__ void __launch_bounds__(256) k_pool(const float* __restrict__ x,
                                              const float* __restrict__ attn,
                                              float* __restrict__ ws) {
    __shared__ __align__(16) float  lx[2][32 * 128];  // 2 x 16 KB fp32, [kk][c]
    __shared__ __align__(16) __bf16 la[2][H_ * 32];   // 2 x 1 KB bf16, [h][kk]
    int t = threadIdx.x;
    int b = blockIdx.x >> 3, cq = blockIdx.x & 7;     // grid 256 = 32 b * 8 col-slices
    int w = t >> 5, lane = t & 31, hi = lane >> 4, nloc = lane & 15;
    const size_t xbase = (size_t)b * L_ * D_ + cq * 128;
    int skk = t >> 3, scb = (t & 7) * 16;             // staging coords: row, 16-col chunk

    auto stage = [&](int buf, int k) {
        const float* src = x + xbase + (size_t)(k + skk) * D_ + scb;
        float* dst = &lx[buf][skk * 128 + scb];
#if ASYNC_LDS
        #pragma unroll
        for (int i = 0; i < 4; i++)
            __builtin_amdgcn_global_load_async_to_lds_b128(
                (GAS v4i*)(src + i*4),
                (LAS v4i*)(dst + i*4), 0, 0);
#else
        #pragma unroll
        for (int i = 0; i < 4; i++)
            *(float4*)(dst + i*4) = *(const float4*)(src + i*4);
#endif
        int h = t >> 4, kk = (t & 15) * 2;            // attn tile 16x32 -> bf16
        size_t ab = (size_t)(b * H_ + h) * L_ + k + kk;
        la[buf][h*32 + kk]     = (__bf16)attn[ab];
        la[buf][h*32 + kk + 1] = (__bf16)attn[ab + 1];
    };

    v8f c0 = {};
    stage(0, 0);
    int cur = 0;
    for (int k = 0; k < L_; k += 32) {
#if ASYNC_LDS
        __builtin_amdgcn_s_wait_asynccnt(0);
#endif
        __syncthreads();                               // buffer `cur` ready for all waves
        if (k + 32 < L_) stage(cur ^ 1, k + 32);       // prefetch next tile into other buffer
        // A fragment: attn rows (M = head)
        const __bf16* ar = &la[cur][nloc * 32];
        v8bf alo = *(const v8bf*)(ar + hi*8);
        v8bf ahi = *(const v8bf*)(ar + 16 + hi*8);
        v16bf av = __builtin_shufflevector(alo, ahi, 0,1,2,3,4,5,6,7,8,9,10,11,12,13,14,15);
        // B fragment: x columns, gathered from fp32 tile and converted
        const float* xc = &lx[cur][0];
        int c = w*16 + nloc;
        v16bf bv;
        #pragma unroll
        for (int j = 0; j < 16; j++) {
            int kk = ((j & 8) << 1) + hi*8 + (j & 7);
            bv[j] = (__bf16)xc[kk*128 + c];
        }
        c0 = __builtin_amdgcn_wmma_f32_16x16x32_bf16(false, av, false, bv, (short)0, c0, false, false);
        cur ^= 1;
    }
    int col0 = cq*128 + w*16 + nloc;
    #pragma unroll
    for (int r = 0; r < 8; r++) {
        int head = r + hi*8;
        ws[WS_POOLEDX + (size_t)(b * H_ + head) * D_ + col0] = c0[r];
    }
}

// ---------------- K4a: ctx[b,j] = Wv[j,:].pooled_x[b,h(j),:] + bv[j] ------
__global__ void __launch_bounds__(256) k_ctx(const float* __restrict__ in_w,
                                             const float* __restrict__ in_b,
                                             float* __restrict__ ws) {
    int t = threadIdx.x;
    int bb = t & 31, rg = t >> 5;          // grid 16 blocks of 64 rows
    for (int jj = 0; jj < 8; jj++) {
        int j = blockIdx.x * 64 + rg * 8 + jj;
        int h = j >> 6;
        const float* wr = in_w + (size_t)(2*D_ + j) * D_;
        const float* px = ws + WS_POOLEDX + (size_t)(bb * H_ + h) * D_;
        float acc = 0.f;
        for (int e = 0; e < D_; e++) acc += wr[e] * px[e];
        ws[WS_CTX + bb * D_ + j] = acc + in_b[2*D_ + j];
    }
}

// ---------------- K4b: y[b,j] = out_w[j,:].ctx[b,:] + out_b[j] ------------
__global__ void __launch_bounds__(256) k_outproj(const float* __restrict__ out_w,
                                                 const float* __restrict__ out_b,
                                                 float* __restrict__ ws) {
    int t = threadIdx.x;
    int bb = t & 31, rg = t >> 5;          // grid 16 blocks of 64 rows
    for (int jj = 0; jj < 8; jj++) {
        int j = blockIdx.x * 64 + rg * 8 + jj;
        const float* wr = out_w + (size_t)j * D_;
        const float* cx = ws + WS_CTX + bb * D_;
        float acc = 0.f;
        for (int e = 0; e < D_; e++) acc += wr[e] * cx[e];
        ws[WS_Y + bb * D_ + j] = acc + out_b[j];
    }
}

// ---------------- K4c: pooled = LayerNorm(y) -> d_out ---------------------
__global__ void __launch_bounds__(256) k_ln_out(const float* __restrict__ ws,
                                                const float* __restrict__ on_w,
                                                const float* __restrict__ on_b,
                                                float* __restrict__ out) {
    __shared__ float red[256];
    int t = threadIdx.x, b = blockIdx.x;   // grid 32
    const float* y = ws + WS_Y + (size_t)b * D_;
    float xl[4], s0 = 0.f, s1 = 0.f;
    #pragma unroll
    for (int i = 0; i < 4; i++) { xl[i] = y[t*4 + i]; s0 += xl[i]; s1 += xl[i]*xl[i]; }
    red[t] = s0; __syncthreads();
    for (int s = 128; s > 0; s >>= 1) { if (t < s) red[t] += red[t+s]; __syncthreads(); }
    float mu = red[0] * (1.0f / D_);
    __syncthreads();
    red[t] = s1; __syncthreads();
    for (int s = 128; s > 0; s >>= 1) { if (t < s) red[t] += red[t+s]; __syncthreads(); }
    float var = red[0] * (1.0f / D_) - mu*mu;
    float rs = rsqrtf(var + 1e-5f);
    #pragma unroll
    for (int i = 0; i < 4; i++) {
        int e = t*4 + i;
        out[(size_t)b * D_ + e] = (xl[i] - mu) * rs * on_w[e] + on_b[e];
    }
}

extern "C" void kernel_launch(void* const* d_in, const int* in_sizes, int n_in,
                              void* d_out, int out_size, void* d_ws, size_t ws_size,
                              hipStream_t stream) {
    const float* ge    = (const float*)d_in[0];
    const float* sym   = (const float*)d_in[1];
    const float* qn_w  = (const float*)d_in[2];
    const float* qn_b  = (const float*)d_in[3];
    const float* in_w  = (const float*)d_in[4];
    const float* in_b  = (const float*)d_in[5];
    const float* out_w = (const float*)d_in[6];
    const float* out_b = (const float*)d_in[7];
    const float* on_w  = (const float*)d_in[8];
    const float* on_b  = (const float*)d_in[9];
    const int*   pos   = (const int*)d_in[10];

    float* ws  = (float*)d_ws;
    float* out = (float*)d_out;
    float* attn_out = out + (size_t)B_ * D_;   // d_out = [pooled (B*D) | attn (B*H*L)]

    k_q0     <<<1,   256, 0, stream>>>(sym, qn_w, qn_b, pos, ws);
    k_qproj  <<<16,  256, 0, stream>>>(in_w, in_b, ws);
    k_wscore <<<64,  256, 0, stream>>>(in_w, ws);
    k_scores <<<B_ * (L_/128), 256, 0, stream>>>(ge, in_b, ws);
    k_softmax<<<B_ * H_, 256, 0, stream>>>(ws, attn_out);
    k_pool   <<<B_ * 8, 256, 0, stream>>>(ge, attn_out, ws);
    k_ctx    <<<16,  256, 0, stream>>>(in_w, in_b, ws);
    k_outproj<<<16,  256, 0, stream>>>(out_w, out_b, ws);
    k_ln_out <<<B_,  256, 0, stream>>>(ws, on_w, on_b, out);
}
